// LLowRankBertSelfAttention_18846316495550
// MI455X (gfx1250) — matrix operand order repaired and checked by reading, not compiled, not run on hardware
//
#include <hip/hip_runtime.h>

typedef __bf16 bf16;
typedef __attribute__((ext_vector_type(8)))  float  v8f;
typedef __attribute__((ext_vector_type(4)))  float  f4;
typedef __attribute__((ext_vector_type(4)))  int    v4i;
typedef __attribute__((ext_vector_type(8)))  __bf16 v8bf;
typedef __attribute__((ext_vector_type(16))) __bf16 v16bf;

union V16u { v16bf v; v8bf h[2]; };
static __device__ __forceinline__ v16bf cat8(v8bf lo, v8bf hi) {
  V16u u; u.h[0] = lo; u.h[1] = hi; return u.v;
}

#define HIDDEN 1024
#define HEADS  16
#define HD     64
#define RANK   256
#define BATCH  4
#define SEQ    2048
#define MTOT   (BATCH*SEQ)

// ---- gfx1250 async global->LDS copy (ASYNCcnt path), with fallback --------
#if __has_builtin(__builtin_amdgcn_global_load_async_to_lds_b128) && \
    __has_builtin(__builtin_amdgcn_s_wait_asynccnt)
#define USE_ASYNC_LDS 1
#else
#define USE_ASYNC_LDS 0
#endif

#if USE_ASYNC_LDS
#define ASYNC_CP_B128(gp, lp)                                                  \
  __builtin_amdgcn_global_load_async_to_lds_b128(                              \
      (__attribute__((address_space(1))) v4i*)(gp),                            \
      (__attribute__((address_space(3))) v4i*)(lp), 0, 0)
#define ASYNC_WAIT() __builtin_amdgcn_s_wait_asynccnt(0)
#endif

// ---- 16-lane butterfly reductions: v_permlane16_b32 (VALU) with fallback --
#if __has_builtin(__builtin_amdgcn_permlane16)
static __device__ __forceinline__ float plx(float v, unsigned lo, unsigned hi) {
  unsigned u = __builtin_bit_cast(unsigned, v);
  unsigned r = __builtin_amdgcn_permlane16(u, u, lo, hi, false, false);
  return __builtin_bit_cast(float, r);
}
static __device__ __forceinline__ float red16_max(float v) {
  v = fmaxf(v, plx(v, 0x67452301u, 0xEFCDAB89u));   // xor 1
  v = fmaxf(v, plx(v, 0x54761032u, 0xDCFE98BAu));   // xor 2
  v = fmaxf(v, plx(v, 0x32107654u, 0xBA98FEDCu));   // xor 4
  v = fmaxf(v, plx(v, 0xFEDCBA98u, 0x76543210u));   // xor 8
  return v;
}
static __device__ __forceinline__ float red16_sum(float v) {
  v += plx(v, 0x67452301u, 0xEFCDAB89u);
  v += plx(v, 0x54761032u, 0xDCFE98BAu);
  v += plx(v, 0x32107654u, 0xBA98FEDCu);
  v += plx(v, 0xFEDCBA98u, 0x76543210u);
  return v;
}
#else
static __device__ __forceinline__ float red16_max(float v) {
  for (int off = 1; off < 16; off <<= 1) v = fmaxf(v, __shfl_xor(v, off, 32));
  return v;
}
static __device__ __forceinline__ float red16_sum(float v) {
  for (int off = 1; off < 16; off <<= 1) v += __shfl_xor(v, off, 32);
  return v;
}
#endif

// ---------------------------------------------------------------------------
// GEMM: out = A[M,K] * W[K,N] (+bias). A is f32 or bf16, W is f32, out bf16.
// Block: 256 threads = 8 waves arranged 4x2; block tile 128x64; k-step 32.
// Each wave computes a 32x32 tile = 2x2 v_wmma_f32_16x16x32_bf16 accumulators.
// LDS pitches padded (40 half-words = 80B = 20 banks) to avoid bank conflicts.
// ---------------------------------------------------------------------------
#define APITCH 40
#define BPITCH 40

template<bool ABF16, bool SCATTER>
__global__ __launch_bounds__(256)
void gemm_bf16(const void* __restrict__ Ap, const float* __restrict__ W,
               const float* __restrict__ bias, bf16* __restrict__ out,
               int M, int K, int N)
{
  __shared__ bf16 a_lds[128 * APITCH];
  __shared__ bf16 b_lds[64 * BPITCH];

  const int tid  = threadIdx.x;
  const int lane = tid & 31;
  const int wave = tid >> 5;
  const int wrow = wave >> 1;     // 0..3
  const int wcol = wave & 1;      // 0..1
  const int ln   = lane & 15;
  const int half = lane >> 4;

  const int m0 = blockIdx.y * 128;
  const int n0 = blockIdx.x * 64;

  v8f acc[2][2] = {};

  const int ar = tid >> 1;         // 0..127  A-tile row
  const int ac = (tid & 1) * 16;   // 0 / 16  A-tile col chunk
  const int bk = tid >> 3;         // 0..31   B-tile k row
  const int bn = (tid & 7) * 8;    // B-tile n chunk

  for (int k0 = 0; k0 < K; k0 += 32) {
    // ---- stage A tile (128 x 32, bf16, row-major) ----
    if (ABF16) {
      const bf16* p = (const bf16*)Ap + (size_t)(m0 + ar) * K + k0 + ac;
#if USE_ASYNC_LDS
      ASYNC_CP_B128(p,     &a_lds[ar * APITCH + ac]);
      ASYNC_CP_B128(p + 8, &a_lds[ar * APITCH + ac + 8]);
#else
      *(v8bf*)&a_lds[ar * APITCH + ac]     = *(const v8bf*)p;
      *(v8bf*)&a_lds[ar * APITCH + ac + 8] = *(const v8bf*)(p + 8);
#endif
    } else {
      const float* p = (const float*)Ap + (size_t)(m0 + ar) * K + k0 + ac;
      f4 f0 = *(const f4*)p, f1 = *(const f4*)(p + 4);
      f4 f2 = *(const f4*)(p + 8), f3 = *(const f4*)(p + 12);
      v8bf x0, x1;
      #pragma unroll
      for (int e = 0; e < 4; e++) {
        x0[e]     = (bf16)f0[e];  x0[4 + e] = (bf16)f1[e];
        x1[e]     = (bf16)f2[e];  x1[4 + e] = (bf16)f3[e];
      }
      *(v8bf*)&a_lds[ar * APITCH + ac]     = x0;
      *(v8bf*)&a_lds[ar * APITCH + ac + 8] = x1;
    }
    // ---- stage B tile transposed: b_lds[n][k] = W[k0+k][n0+n] ----
    {
      const float* p = W + (size_t)(k0 + bk) * N + n0 + bn;
      f4 f0 = *(const f4*)p, f1 = *(const f4*)(p + 4);
      #pragma unroll
      for (int e = 0; e < 4; e++) {
        b_lds[(bn + e)     * BPITCH + bk] = (bf16)f0[e];
        b_lds[(bn + 4 + e) * BPITCH + bk] = (bf16)f1[e];
      }
    }
#if USE_ASYNC_LDS
    if (ABF16) ASYNC_WAIT();
#endif
    __syncthreads();

    // ---- fragments + WMMA ----
    v16bf afr[2], bfr[2];
    #pragma unroll
    for (int i = 0; i < 2; i++) {
      // A layout: lane holds row (lane%16); element j -> k = 16*(j/8)+(j%8)+8*half
      const bf16* p = &a_lds[(wrow * 32 + i * 16 + ln) * APITCH + 8 * half];
      afr[i] = cat8(*(const v8bf*)p, *(const v8bf*)(p + 16));
    }
    #pragma unroll
    for (int j = 0; j < 2; j++) {
      // B layout: lane holds col (lane%16); element j -> k = 16*(j/8)+(j%8)+8*half
      const bf16* p = &b_lds[(wcol * 32 + j * 16 + ln) * BPITCH + 8 * half];
      bfr[j] = cat8(*(const v8bf*)p, *(const v8bf*)(p + 16));
    }
    #pragma unroll
    for (int i = 0; i < 2; i++)
      #pragma unroll
      for (int j = 0; j < 2; j++)
        acc[i][j] = __builtin_amdgcn_wmma_f32_16x16x32_bf16(
            false, afr[i], false, bfr[j], (short)0, acc[i][j], false, false);

    __syncthreads();
  }

  // ---- epilogue: C/D layout row = r + 8*half, col = lane%16 ----
  #pragma unroll
  for (int i = 0; i < 2; i++)
    #pragma unroll
    for (int j = 0; j < 2; j++)
      #pragma unroll
      for (int r = 0; r < 8; r++) {
        int row = m0 + wrow * 32 + i * 16 + r + 8 * half;
        int col = n0 + wcol * 32 + j * 16 + ln;
        float v = acc[i][j][r];
        if (SCATTER) {
          v += bias[col];
          int b = row >> 11, s = row & (SEQ - 1);
          int h = col >> 6,  d = col & (HD - 1);
          out[(((size_t)b * HEADS + h) * SEQ + s) * HD + d] = (bf16)v;
        } else {
          out[(size_t)row * N + col] = (bf16)v;
        }
      }
}

// ---------------------------------------------------------------------------
// Flash attention: grid (S/64, H, B), block 128 = 4 waves; each wave owns
// 16 query rows; iterate key tiles of 32 with online softmax (f32 state).
// Scores and PV both via v_wmma_f32_16x16x32_bf16. K tile staged with
// global_load_async_to_lds_b128 when available.
// ---------------------------------------------------------------------------
#define KPITCH 72
#define VPITCH 40
#define PPITCH 40

__global__ __launch_bounds__(128)
void attn_fa(const bf16* __restrict__ Q, const bf16* __restrict__ Kx,
             const bf16* __restrict__ V, float* __restrict__ out)
{
  __shared__ bf16 k_lds[32 * KPITCH];        // K tile [key][d]
  __shared__ bf16 vT[64 * VPITCH];           // V tile transposed [d][key]
  __shared__ bf16 p_lds[4][16 * PPITCH];     // per-wave P tile [qrow][key]

  const int tid  = threadIdx.x;
  const int lane = tid & 31, wave = tid >> 5;
  const int ln   = lane & 15, half = lane >> 4;

  const int qt = blockIdx.x, h = blockIdx.y, b = blockIdx.z;
  const size_t headbase = ((size_t)b * HEADS + h) * SEQ * HD;

  // ---- Q fragments for this wave's 16 rows (2 d-steps of 32) ----
  v16bf qf[2];
  {
    const bf16* qp = Q + headbase + (size_t)(qt * 64 + wave * 16 + ln) * HD;
    #pragma unroll
    for (int ds = 0; ds < 2; ds++)
      qf[ds] = cat8(*(const v8bf*)(qp + ds * 32 + 8 * half),
                    *(const v8bf*)(qp + ds * 32 + 16 + 8 * half));
  }

  float m_i[8], l_i[8];
  #pragma unroll
  for (int r = 0; r < 8; r++) { m_i[r] = -__builtin_inff(); l_i[r] = 0.f; }
  v8f acc_o[4] = {};

  const int lr = tid >> 2;          // 0..31 staging row
  const int lc = (tid & 3) * 16;    // staging d-chunk

  for (int kt = 0; kt < SEQ / 32; kt++) {
    // ---- stage K tile row-major (async), V tile transposed (manual) ----
    {
      const bf16* kp = Kx + headbase + (size_t)(kt * 32 + lr) * HD + lc;
#if USE_ASYNC_LDS
      ASYNC_CP_B128(kp,     &k_lds[lr * KPITCH + lc]);
      ASYNC_CP_B128(kp + 8, &k_lds[lr * KPITCH + lc + 8]);
#else
      *(v8bf*)&k_lds[lr * KPITCH + lc]     = *(const v8bf*)kp;
      *(v8bf*)&k_lds[lr * KPITCH + lc + 8] = *(const v8bf*)(kp + 8);
#endif
      const bf16* vp = V + headbase + (size_t)(kt * 32 + lr) * HD + lc;
      v8bf y0 = *(const v8bf*)vp, y1 = *(const v8bf*)(vp + 8);
      #pragma unroll
      for (int e = 0; e < 8; e++) {
        vT[(lc + e)     * VPITCH + lr] = y0[e];
        vT[(lc + 8 + e) * VPITCH + lr] = y1[e];
      }
      if (kt + 1 < SEQ / 32) {   // global_prefetch_b8 for next tiles
        __builtin_prefetch(Kx + headbase + (size_t)((kt + 1) * 32 + lr) * HD + lc, 0, 3);
        __builtin_prefetch(V  + headbase + (size_t)((kt + 1) * 32 + lr) * HD + lc, 0, 3);
      }
    }
#if USE_ASYNC_LDS
    ASYNC_WAIT();
#endif
    __syncthreads();

    // ---- scores: S = Q(16x64) * K^T(64x32) ----
    v8f sc[2] = {};
    #pragma unroll
    for (int ds = 0; ds < 2; ds++)
      #pragma unroll
      for (int j = 0; j < 2; j++) {
        const bf16* p = &k_lds[(j * 16 + ln) * KPITCH + ds * 32 + 8 * half];
        v16bf kf = cat8(*(const v8bf*)p, *(const v8bf*)(p + 16));
        sc[j] = __builtin_amdgcn_wmma_f32_16x16x32_bf16(
            false, qf[ds], false, kf, (short)0, sc[j], false, false);
      }

    // ---- online softmax (row = r + 8*half spread over 16 lanes/half) ----
    float p0[8], p1[8];
    #pragma unroll
    for (int r = 0; r < 8; r++) {
      float a0 = sc[0][r] * 0.125f, a1 = sc[1][r] * 0.125f;   // 1/sqrt(64)
      float nm = fmaxf(m_i[r], red16_max(fmaxf(a0, a1)));
      float e0 = __expf(a0 - nm), e1 = __expf(a1 - nm);
      float rs = red16_sum(e0 + e1);
      float alpha = __expf(m_i[r] - nm);
      l_i[r] = l_i[r] * alpha + rs;
      m_i[r] = nm;
      #pragma unroll
      for (int t = 0; t < 4; t++) acc_o[t][r] *= alpha;
      p0[r] = e0; p1[r] = e1;
    }
    // ---- stage P (C-layout -> A-layout via LDS) ----
    #pragma unroll
    for (int r = 0; r < 8; r++) {
      p_lds[wave][(r + 8 * half) * PPITCH + ln]      = (bf16)p0[r];
      p_lds[wave][(r + 8 * half) * PPITCH + 16 + ln] = (bf16)p1[r];
    }
    __syncthreads();

    // ---- PV: O += P(16x32) * V(32x64) ----
    {
      const bf16* pr = &p_lds[wave][ln * PPITCH + 8 * half];
      v16bf pf = cat8(*(const v8bf*)pr, *(const v8bf*)(pr + 16));
      #pragma unroll
      for (int t = 0; t < 4; t++) {
        const bf16* vp = &vT[(t * 16 + ln) * VPITCH + 8 * half];
        v16bf vf = cat8(*(const v8bf*)vp, *(const v8bf*)(vp + 16));
        acc_o[t] = __builtin_amdgcn_wmma_f32_16x16x32_bf16(
            false, pf, false, vf, (short)0, acc_o[t], false, false);
      }
    }
    __syncthreads();
  }

  // ---- normalize + write ctx [B,S,HIDDEN] f32 ----
  #pragma unroll
  for (int t = 0; t < 4; t++)
    #pragma unroll
    for (int r = 0; r < 8; r++) {
      int s_idx = qt * 64 + wave * 16 + r + 8 * half;
      int col   = h * HD + t * 16 + ln;
      out[((size_t)b * SEQ + s_idx) * HIDDEN + col] = acc_o[t][r] / l_i[r];
    }
}

// ---------------------------------------------------------------------------
extern "C" void kernel_launch(void* const* d_in, const int* in_sizes, int n_in,
                              void* d_out, int out_size, void* d_ws, size_t ws_size,
                              hipStream_t stream) {
  const float* hidden  = (const float*)d_in[0];
  const float* Aw[3]   = {(const float*)d_in[1], (const float*)d_in[4], (const float*)d_in[7]};
  const float* Bw[3]   = {(const float*)d_in[2], (const float*)d_in[5], (const float*)d_in[8]};
  const float* bias[3] = {(const float*)d_in[3], (const float*)d_in[6], (const float*)d_in[9]};

  // workspace: XA (reused) + bf16 Q,K,V in [B,H,S,D]
  bf16* xa = (bf16*)d_ws;
  bf16* Qb = xa + (size_t)MTOT * RANK;
  bf16* Kb = Qb + (size_t)MTOT * HIDDEN;
  bf16* Vb = Kb + (size_t)MTOT * HIDDEN;
  bf16* dst[3] = {Qb, Kb, Vb};

  for (int i = 0; i < 3; i++) {
    gemm_bf16<false, false><<<dim3(RANK / 64, MTOT / 128), 256, 0, stream>>>(
        hidden, Aw[i], nullptr, xa, MTOT, HIDDEN, RANK);
    gemm_bf16<true, true><<<dim3(HIDDEN / 64, MTOT / 128), 256, 0, stream>>>(
        xa, Bw[i], bias[i], dst[i], MTOT, RANK, HIDDEN);
  }
  attn_fa<<<dim3(SEQ / 64, HEADS, BATCH), 128, 0, stream>>>(
      Qb, Kb, Vb, (float*)d_out);
}
